// FastLSH_74225624809851
// MI455X (gfx1250) — compile-verified, matrix operand order checked by reading, not run on hardware
//
#include <hip/hip_runtime.h>

#define B_ 16
#define S_ 4096
#define D_ 512
#define M_ 128
#define K_ 16

typedef __bf16 v16bf __attribute__((ext_vector_type(16)));
typedef float  v8f   __attribute__((ext_vector_type(8)));
typedef __bf16 v4bf  __attribute__((ext_vector_type(4)));
typedef int    vi4   __attribute__((vector_size(16)));   // matches builtin param type

typedef __attribute__((address_space(1))) vi4* gvi4_p;
typedef __attribute__((address_space(3))) vi4* lvi4_p;

// ---- CDNA5 async global->LDS copy (ASYNCcnt-tracked), builtin or asm fallback
__device__ __forceinline__ void async_copy_b128(const void* gsrc, void* lds_dst) {
#if __has_builtin(__builtin_amdgcn_global_load_async_to_lds_b128)
    __builtin_amdgcn_global_load_async_to_lds_b128(
        (gvi4_p)(void*)gsrc, (lvi4_p)lds_dst, 0, 0);
#else
    asm volatile("global_load_async_to_lds_b128 %0, %1, off"
                 :: "v"((unsigned)(unsigned long long)lds_dst), "v"(gsrc)
                 : "memory");
#endif
}

#if __has_builtin(__builtin_amdgcn_s_wait_asynccnt)
#define WAIT_ASYNC(n) __builtin_amdgcn_s_wait_asynccnt(n)
#else
#define WAIT_ASYNC(n) asm volatile("s_wait_asynccnt %0" :: "i"(n) : "memory")
#endif

// ---------------- Kernel 1: gather sampled rows, convert to bf16, compute y2 ----
__global__ void lsh_gather(const float* __restrict__ emb,
                           const int* __restrict__ idx,
                           __bf16* __restrict__ samp,
                           float* __restrict__ y2) {
    int b = blockIdx.x >> 7;      // /128
    int m = blockIdx.x & 127;
    int t = threadIdx.x;          // 0..127, each handles 4 floats
    int s = idx[m];

    const float4* src = (const float4*)(emb + ((size_t)b * S_ + s) * D_);
    float4 f = src[t];

    v4bf h;
    h[0] = (__bf16)f.x; h[1] = (__bf16)f.y; h[2] = (__bf16)f.z; h[3] = (__bf16)f.w;
    *(v4bf*)(samp + ((size_t)b * M_ + m) * D_ + t * 4) = h;

    float ss = f.x*f.x + f.y*f.y + f.z*f.z + f.w*f.w;
    #pragma unroll
    for (int off = 16; off >= 1; off >>= 1) ss += __shfl_xor(ss, off);

    __shared__ float red[4];
    int lane = t & 31, wid = t >> 5;
    if (lane == 0) red[wid] = ss;
    __syncthreads();
    if (t == 0) y2[b * M_ + m] = red[0] + red[1] + red[2] + red[3];
}

// ---------------- Kernel 2: fused GEMM (bf16 WMMA) + distance + top-16 ---------
__global__ void __launch_bounds__(256)
lsh_main(const float* __restrict__ emb,
         const int* __restrict__ idx,
         const __bf16* __restrict__ samp,
         const float* __restrict__ y2g,
         int* __restrict__ out_nbr,
         float* __restrict__ out_dst) {
    __shared__ __bf16 Bs[2][M_][32];       // 16 KB : double-buffered B k-slice
    __shared__ float  dist_s[8][M_][8];    // 32 KB : per-wave 8-row distance tile
    __shared__ float  y2_s[M_];
    __shared__ float  x2_s[8][16];
    __shared__ int    idx_s[M_];

    const int tid  = threadIdx.x;
    const int lane = tid & 31;
    const int w    = tid >> 5;             // wave 0..7
    const int b    = blockIdx.y;
    const int s0   = blockIdx.x * 128;     // row tile base
    const int half = lane >> 4;            // 0: k-half 0, 1: k-half 8
    const int khalf = half * 8;
    const int rlane = lane & 15;

    if (tid < M_) {
        y2_s[tid]  = y2g[b * M_ + tid];
        idx_s[tid] = idx[tid];
    }

    const int row = s0 + 16 * w + rlane;
    const float* arow = emb + ((size_t)b * S_ + row) * D_ + khalf;

    v8f acc[8];
    #pragma unroll
    for (int t = 0; t < 8; ++t) acc[t] = (v8f)0.0f;
    float x2p = 0.0f;

    // async-stage one 128x32 bf16 slice of sampled into Bs[buf]
    const int sm = tid >> 1, shh = tid & 1;
    const __bf16* sbase = samp + ((size_t)b * M_ + sm) * D_ + 16 * shh;
    auto issue_slice = [&](int kk, int buf) {
        const __bf16* gsrc = sbase + kk;
        async_copy_b128(gsrc,     &Bs[buf][sm][16 * shh]);
        async_copy_b128(gsrc + 8, &Bs[buf][sm][16 * shh + 8]);
    };

    issue_slice(0, 0);
    for (int kk = 0, it = 0; kk < D_; kk += 32, ++it) {
        const int cur = it & 1;
        const bool more = (kk + 32 < D_);
        if (more) {
            issue_slice(kk + 32, cur ^ 1);      // prefetch next slice into other buffer
            WAIT_ASYNC(2);                      // in-order: slice `cur` is resident
            __builtin_prefetch(arow + kk + 32, 0, 3);
        } else {
            WAIT_ASYNC(0);
        }
        __syncthreads();                        // all waves' slice-cur data visible

        // A fragment: two contiguous 8-float runs per lane (documented 16-bit A layout)
        float4 f0 = *(const float4*)(arow + kk);
        float4 f1 = *(const float4*)(arow + kk + 4);
        float4 f2 = *(const float4*)(arow + kk + 16);
        float4 f3 = *(const float4*)(arow + kk + 20);
        x2p += f0.x*f0.x + f0.y*f0.y + f0.z*f0.z + f0.w*f0.w
             + f1.x*f1.x + f1.y*f1.y + f1.z*f1.z + f1.w*f1.w
             + f2.x*f2.x + f2.y*f2.y + f2.z*f2.z + f2.w*f2.w
             + f3.x*f3.x + f3.y*f3.y + f3.z*f3.z + f3.w*f3.w;

        v16bf a;
        a[0]  = (__bf16)f0.x; a[1]  = (__bf16)f0.y; a[2]  = (__bf16)f0.z; a[3]  = (__bf16)f0.w;
        a[4]  = (__bf16)f1.x; a[5]  = (__bf16)f1.y; a[6]  = (__bf16)f1.z; a[7]  = (__bf16)f1.w;
        a[8]  = (__bf16)f2.x; a[9]  = (__bf16)f2.y; a[10] = (__bf16)f2.z; a[11] = (__bf16)f2.w;
        a[12] = (__bf16)f3.x; a[13] = (__bf16)f3.y; a[14] = (__bf16)f3.z; a[15] = (__bf16)f3.w;

        #pragma unroll
        for (int t = 0; t < 8; ++t) {
            int m = 16 * t + rlane;
            union { uint4 u[2]; v16bf v; } bb;
            bb.u[0] = *(const uint4*)&Bs[cur][m][khalf];
            bb.u[1] = *(const uint4*)&Bs[cur][m][16 + khalf];
            acc[t] = __builtin_amdgcn_wmma_f32_16x16x32_bf16(
                false, a, false, bb.v, (short)0, acc[t], false, false);
        }
        __syncthreads();                        // reads done before buffer reuse
    }

    // x2 for full row: combine the two k-halves (lanes L and L^16 cover row L%16)
    float x2full = x2p + __shfl_xor(x2p, 16);
    if (lane < 16) x2_s[w][lane] = x2full;

    // ---- distance + top-16, 8 rows per pass (pass == lane-half owning the rows)
    for (int pass = 0; pass < 2; ++pass) {
        __syncthreads();
        if (half == pass) {
            #pragma unroll
            for (int t = 0; t < 8; ++t) {
                int col = 16 * t + rlane;
                float yv = y2_s[col];
                #pragma unroll
                for (int r = 0; r < 8; ++r) {
                    int rowl = r + 8 * half;
                    float d2 = x2_s[w][rowl] + yv - 2.0f * acc[t][r];
                    dist_s[w][col][r] = sqrtf(fmaxf(d2, 0.0f));
                }
            }
        }
        __syncthreads();

        for (int q = 0; q < 8; ++q) {
            unsigned long long key[4];
            #pragma unroll
            for (int j = 0; j < 4; ++j) {
                int c = lane + 32 * j;
                float v = dist_s[w][c][q];
                key[j] = ((unsigned long long)__float_as_uint(v) << 32) | (unsigned)c;
            }
            float res_v = 0.0f; int res_c = 0;
            for (int i = 0; i < K_; ++i) {
                unsigned long long km = key[0];
                if (key[1] < km) km = key[1];
                if (key[2] < km) km = key[2];
                if (key[3] < km) km = key[3];
                #pragma unroll
                for (int off = 16; off >= 1; off >>= 1) {
                    unsigned long long o = __shfl_xor(km, off);
                    if (o < km) km = o;
                }
                int   mc = (int)(km & 0xffffffffu);
                float mv = __uint_as_float((unsigned)(km >> 32));
                if (lane == i) { res_v = mv; res_c = mc; }
                #pragma unroll
                for (int j = 0; j < 4; ++j)
                    if ((int)(key[j] & 0xffffffffu) == mc) key[j] = ~0ull;
            }
            int s_row = s0 + 16 * w + pass * 8 + q;
            size_t obase = ((size_t)b * S_ + s_row) * K_;
            if (lane < K_) {
                out_dst[obase + lane] = res_v;
                out_nbr[obase + lane] = idx_s[res_c];
            }
        }
    }
}

extern "C" void kernel_launch(void* const* d_in, const int* in_sizes, int n_in,
                              void* d_out, int out_size, void* d_ws, size_t ws_size,
                              hipStream_t stream) {
    const float* emb = (const float*)d_in[0];
    const int*   idx = (const int*)d_in[1];
    // d_in[2] is k (==16), hardcoded as K_.

    __bf16* samp = (__bf16*)d_ws;
    float*  y2   = (float*)((char*)d_ws + (size_t)B_ * M_ * D_ * sizeof(__bf16));

    int*   out_nbr = (int*)d_out;
    float* out_dst = (float*)d_out + (size_t)B_ * S_ * K_;

    lsh_gather<<<dim3(B_ * M_), 128, 0, stream>>>(emb, idx, samp, y2);
    lsh_main<<<dim3(S_ / 128, B_), 256, 0, stream>>>(emb, idx, samp, y2, out_nbr, out_dst);
}
